// GPT_64149631533745
// MI455X (gfx1250) — compile-verified, matrix operand order checked by reading, not run on hardware
//
#include <hip/hip_runtime.h>
#include <hip/hip_bf16.h>

// ---------------------------------------------------------------------------
// Types / helpers
// ---------------------------------------------------------------------------
typedef __attribute__((ext_vector_type(16))) __bf16 v16bf;
typedef __attribute__((ext_vector_type(8)))  float  v8f;

#define DMODEL 512
#define NHEADS 2
#define HDIM   256
#define FFDIM  2048
#define SEQ    1024
#define BATCH  8
#define MTOK   (BATCH * SEQ)        // 8192 tokens
#define NEGINF (-3.0e38f)

// TDM availability / flavor detection (compile-time, consistent host+device)
#if __has_builtin(__builtin_amdgcn_tensor_load_to_lds)
#define USE_TDM 1
#else
#define USE_TDM 0
#endif
#if __has_include(<hip/amd_detail/amd_gfx1250_TDM.h>)
#define TDM_SIX_ARGS 1              // amdgpu-toolchain (clang-23): 6-arg builtin
#else
#define TDM_SIX_ARGS 0              // ROCm 7.2 (clang-22): 5-arg builtin
#endif

__device__ __forceinline__ unsigned short f2bf(float f) {
  unsigned int u = __float_as_uint(f);
  u += 0x7FFFu + ((u >> 16) & 1u);          // round-to-nearest-even
  return (unsigned short)(u >> 16);
}

__device__ __forceinline__ v8f wmma_bf16(v16bf a, v16bf b, v8f c) {
  // D = A(16x32 bf16) x B(32x16 bf16) + C(16x16 f32)
  return __builtin_amdgcn_wmma_f32_16x16x32_bf16(false, a, false, b,
                                                 (short)0, c, false, false);
}

union Frag { v16bf v; uint4 q[2]; };

#if USE_TDM
typedef __attribute__((ext_vector_type(4))) unsigned int v4u32;
typedef __attribute__((ext_vector_type(8))) int v8i32;
typedef __attribute__((ext_vector_type(4))) int v4i32;

// Issue one TDM load of a 128x32-element bf16 tile (row-major, row stride
// Kelems) into LDS laid out as [128][40] u16 (64B data + 16B pad per row).
// D# fields per CDNA5 ISA 8.3/8.4: count=1, type=2("image"), data_size=2B,
// pad_interval=16 DWORDs, pad_amount=4 DWORDs.
__device__ __forceinline__ void tdm_load_tile_128x32(
    const unsigned short* gtile, unsigned lds_byte_off,
    int rows_total, int Kelems) {
  unsigned long long ga = (unsigned long long)(size_t)gtile;
  v4u32 g0;
  g0.x = 1u;                                    // count=1, user-mode
  g0.y = lds_byte_off;                          // lds_addr
  g0.z = (unsigned)ga;                          // global_addr[31:0]
  g0.w = (unsigned)((ga >> 32) & 0x01FFFFFFu)   // global_addr[56:32]
         | (2u << 30);                          // type = 2
  v8i32 g1;
  g1[0] = (1 << 16)                             // data_size = 2 bytes
        | (1 << 20)                             // pad_enable
        | (3 << 22)                             // pad_interval: 16 DWORDs
        | (3 << 25);                            // pad_amount:   4 DWORDs
  g1[1] = (Kelems & 0xFFFF) << 16;              // tensor_dim0[15:0]
  g1[2] = (Kelems >> 16) | ((rows_total & 0xFFFF) << 16); // dim0 hi | dim1 lo
  g1[3] = (rows_total >> 16) | (32 << 16);      // dim1 hi | tile_dim0 = 32
  g1[4] = 128;                                  // tile_dim1 = 128, tile_dim2 = 0
  g1[5] = Kelems;                               // tensor_dim0_stride[31:0]
  g1[6] = 0;                                    // stride0 hi | stride1 lo
  g1[7] = 0;                                    // stride1 hi
  v4i32 z4 = {0, 0, 0, 0};
#if TDM_SIX_ARGS
  v8i32 z8 = {0, 0, 0, 0, 0, 0, 0, 0};
  __builtin_amdgcn_tensor_load_to_lds(g0, g1, z4, z4, z8, 0);
#else
  __builtin_amdgcn_tensor_load_to_lds(g0, g1, z4, z4, 0);
#endif
}
#endif  // USE_TDM

// ---------------------------------------------------------------------------
// f32 [K,N] -> bf16 [N,K] transposed conversion (weights), 32x32 LDS tiles.
// ---------------------------------------------------------------------------
__global__ __launch_bounds__(256) void cvtT_kernel(const float* __restrict__ in,
                                                   unsigned short* __restrict__ out,
                                                   int K, int N) {
  __shared__ float tile[32][33];
  int kb = blockIdx.x * 32, nb = blockIdx.y * 32;
  int tx = threadIdx.x & 31, ty = threadIdx.x >> 5;   // ty: 0..7
#pragma unroll
  for (int j = 0; j < 32; j += 8)
    tile[ty + j][tx] = in[(size_t)(kb + ty + j) * N + nb + tx];
  __syncthreads();
#pragma unroll
  for (int j = 0; j < 32; j += 8)
    out[(size_t)(nb + ty + j) * K + kb + tx] = f2bf(tile[tx][ty + j]);
}

// ---------------------------------------------------------------------------
// Embedding: x[t,:] = tok[ids[t],:] + pos[t % SEQ,:]
// ---------------------------------------------------------------------------
__global__ void embed_kernel(const float* __restrict__ tok,
                             const float* __restrict__ pos,
                             const int* __restrict__ ids,
                             float* __restrict__ xf,
                             unsigned short* __restrict__ xb) {
  int t = blockIdx.x;
  int s = t & (SEQ - 1);
  int id = ids[t];
  const float* tr = tok + (size_t)id * DMODEL;
  const float* pr = pos + (size_t)s  * DMODEL;
  for (int c = threadIdx.x; c < DMODEL; c += blockDim.x) {
    float v = tr[c] + pr[c];
    xf[(size_t)t * DMODEL + c] = v;
    xb[(size_t)t * DMODEL + c] = f2bf(v);
  }
}

// ---------------------------------------------------------------------------
// GEMM: C[M,N] = A[M,K](bf16,row-major) * BT[N,K](bf16, pre-transposed)
//        + bias, optional relu, f32/bf16 outs, optional QKV scatter.
// Block tile 128x128, 8 waves, wave tile 32x64 (2x4 WMMA accumulators),
// K-step 32, double-buffered LDS, TDM DMA staging (fallback: reg prefetch).
// ---------------------------------------------------------------------------
__global__ __launch_bounds__(256) void gemm_bf16_kernel(
    const unsigned short* __restrict__ A,
    const unsigned short* __restrict__ BT,
    const float* __restrict__ bias,
    float* __restrict__ C,
    unsigned short* __restrict__ Cbf,
    unsigned short* __restrict__ Qo,
    unsigned short* __restrict__ Ko,
    unsigned short* __restrict__ Vo,
    int M, int N, int K, int relu) {
  __shared__ unsigned short As[2][128][40];   // double-buffered, 80B rows
  __shared__ unsigned short Bs[2][128][40];

  const int tid  = threadIdx.x;
  const int lane = tid & 31;
  const int wid  = tid >> 5;                  // 0..7
  const int wm   = wid & 3;                   // 4 waves along M (32 rows each)
  const int wn   = wid >> 2;                  // 2 waves along N (64 cols each)
  const int bm   = blockIdx.x * 128;
  const int bn   = blockIdx.y * 128;
  const int hi   = lane >> 4;
  const int lo   = lane & 15;
  const int kb   = hi * 8;

  v8f acc[2][4];
#pragma unroll
  for (int i = 0; i < 2; ++i)
#pragma unroll
    for (int j = 0; j < 4; ++j) acc[i][j] = (v8f){0.f,0.f,0.f,0.f,0.f,0.f,0.f,0.f};

  const int steps = K >> 5;

#if USE_TDM
  // ---- TDM double-buffered pipeline: wave 0 drives the DMA engine ----
  if (wid == 0) {
    tdm_load_tile_128x32(A  + (size_t)bm * K, (unsigned)(size_t)&As[0][0][0], M, K);
    tdm_load_tile_128x32(BT + (size_t)bn * K, (unsigned)(size_t)&Bs[0][0][0], N, K);
  }
  for (int s = 0; s < steps; ++s) {
    const int cur = s & 1;
    __syncthreads();                          // readers of buf[cur^1] finished
    if (wid == 0) {
      if (s + 1 < steps) {
        tdm_load_tile_128x32(A  + (size_t)bm * K + (s + 1) * 32,
                             (unsigned)(size_t)&As[cur ^ 1][0][0], M, K);
        tdm_load_tile_128x32(BT + (size_t)bn * K + (s + 1) * 32,
                             (unsigned)(size_t)&Bs[cur ^ 1][0][0], N, K);
        __builtin_amdgcn_s_wait_tensorcnt(2); // tile s done; s+1 in flight
      } else {
        __builtin_amdgcn_s_wait_tensorcnt(0);
      }
    }
    __syncthreads();                          // tile s visible to all waves

    Frag af[2], bf[4];
#pragma unroll
    for (int i = 0; i < 2; ++i) {
      int m = wm * 32 + i * 16 + lo;
      af[i].q[0] = *(const uint4*)(&As[cur][m][kb]);
      af[i].q[1] = *(const uint4*)(&As[cur][m][16 + kb]);
    }
#pragma unroll
    for (int j = 0; j < 4; ++j) {
      int n = wn * 64 + j * 16 + lo;
      bf[j].q[0] = *(const uint4*)(&Bs[cur][n][kb]);
      bf[j].q[1] = *(const uint4*)(&Bs[cur][n][16 + kb]);
    }
#pragma unroll
    for (int i = 0; i < 2; ++i)
#pragma unroll
      for (int j = 0; j < 4; ++j)
        acc[i][j] = wmma_bf16(af[i].v, bf[j].v, acc[i][j]);
  }
#else
  // ---- fallback: register-prefetch, double-buffered LDS, 1 barrier/step ----
  const int srow = tid >> 1;                  // 0..127
  const int scol = (tid & 1) * 16;            // 0 or 16
  const unsigned short* Ap = A  + (size_t)(bm + srow) * K + scol;
  const unsigned short* Bp = BT + (size_t)(bn + srow) * K + scol;

  uint4 pa0 = *(const uint4*)(Ap);
  uint4 pa1 = *(const uint4*)(Ap + 8);
  uint4 pb0 = *(const uint4*)(Bp);
  uint4 pb1 = *(const uint4*)(Bp + 8);
  *(uint4*)(&As[0][srow][scol])     = pa0;
  *(uint4*)(&As[0][srow][scol + 8]) = pa1;
  *(uint4*)(&Bs[0][srow][scol])     = pb0;
  *(uint4*)(&Bs[0][srow][scol + 8]) = pb1;

  for (int s = 0; s < steps; ++s) {
    const int cur = s & 1;
    __syncthreads();                          // buf[cur] ready for everyone
    if (s + 1 < steps) {                      // prefetch next K tile
      Ap += 32; Bp += 32;
      pa0 = *(const uint4*)(Ap);
      pa1 = *(const uint4*)(Ap + 8);
      pb0 = *(const uint4*)(Bp);
      pb1 = *(const uint4*)(Bp + 8);
    }

    Frag af[2], bf[4];
#pragma unroll
    for (int i = 0; i < 2; ++i) {
      int m = wm * 32 + i * 16 + lo;
      af[i].q[0] = *(const uint4*)(&As[cur][m][kb]);
      af[i].q[1] = *(const uint4*)(&As[cur][m][16 + kb]);
    }
#pragma unroll
    for (int j = 0; j < 4; ++j) {
      int n = wn * 64 + j * 16 + lo;
      bf[j].q[0] = *(const uint4*)(&Bs[cur][n][kb]);
      bf[j].q[1] = *(const uint4*)(&Bs[cur][n][16 + kb]);
    }
#pragma unroll
    for (int i = 0; i < 2; ++i)
#pragma unroll
      for (int j = 0; j < 4; ++j)
        acc[i][j] = wmma_bf16(af[i].v, bf[j].v, acc[i][j]);

    if (s + 1 < steps) {                      // store into the idle buffer
      *(uint4*)(&As[cur ^ 1][srow][scol])     = pa0;
      *(uint4*)(&As[cur ^ 1][srow][scol + 8]) = pa1;
      *(uint4*)(&Bs[cur ^ 1][srow][scol])     = pb0;
      *(uint4*)(&Bs[cur ^ 1][srow][scol + 8]) = pb1;
    }
  }
#endif

  // ---- epilogue ----
#pragma unroll
  for (int i = 0; i < 2; ++i) {
#pragma unroll
    for (int j = 0; j < 4; ++j) {
#pragma unroll
      for (int r = 0; r < 8; ++r) {
        int m = bm + wm * 32 + i * 16 + hi * 8 + r;
        int n = bn + wn * 64 + j * 16 + lo;
        float v = acc[i][j][r];
        if (bias) v += bias[n];
        if (relu) v = fmaxf(v, 0.f);
        if (C)   C[(size_t)m * N + n] = v;
        if (Cbf) Cbf[(size_t)m * N + n] = f2bf(v);
        if (Qo) {                             // QKV scatter, N == 3*DMODEL
          int bb = m >> 10;                   // / SEQ
          int ss = m & (SEQ - 1);
          unsigned short bv = f2bf(v);
          int c = n;
          if (c < DMODEL) {                   // Q: [B,H,S,DH]
            int h = c >> 8, dh = c & 255;
            Qo[((((size_t)bb * NHEADS + h) * SEQ + ss) << 8) + dh] = bv;
          } else if (c < 2 * DMODEL) {        // K: [B,H,S,DH]
            c -= DMODEL;
            int h = c >> 8, dh = c & 255;
            Ko[((((size_t)bb * NHEADS + h) * SEQ + ss) << 8) + dh] = bv;
          } else {                            // V: [B,H,DH,S]  (pre-transposed)
            c -= 2 * DMODEL;
            int h = c >> 8, dh = c & 255;
            Vo[(((size_t)bb * NHEADS + h) * HDIM + dh) * SEQ + ss] = bv;
          }
        }
      }
    }
  }
}

// ---------------------------------------------------------------------------
// Flash attention: Q,K in [B,H,S,HDIM] bf16; V in [B,H,HDIM,S] bf16
// (pre-transposed). Output concat-heads [B,S,DMODEL] bf16 (+opt f32).
// Block: 8 waves x 16 q rows = 128 rows; key tiles of 32; online softmax.
// ---------------------------------------------------------------------------
__global__ __launch_bounds__(256) void attn_kernel(
    const unsigned short* __restrict__ Q,
    const unsigned short* __restrict__ Km,
    const unsigned short* __restrict__ Vm,
    float* __restrict__ Of,
    unsigned short* __restrict__ Ob,
    int causal) {
  const int bh  = blockIdx.x;                 // b*H + h
  const int h   = bh & (NHEADS - 1);
  const int b   = bh >> 1;
  const int q0  = blockIdx.y * 128;
  const int tid = threadIdx.x;
  const int lane = tid & 31;
  const int wid  = tid >> 5;                  // 0..7
  const int hi   = lane >> 4;
  const int lo   = lane & 15;
  const int kb   = hi * 8;
  const int qw   = q0 + wid * 16;             // wave's query-row base

  const unsigned short* Qb = Q  + (size_t)bh * SEQ * HDIM;
  const unsigned short* Kb = Km + (size_t)bh * SEQ * HDIM;
  const unsigned short* Vb = Vm + (size_t)bh * HDIM * SEQ;  // [dh][s]

  __shared__ unsigned short Ks[32][HDIM + 8]; // K tile row-major [k][dh]
  __shared__ unsigned short Vs[HDIM][40];     // V tile [dh][k]
  __shared__ unsigned short Ps[8][16][32];    // per-wave probability tile

  v8f oacc[16];
#pragma unroll
  for (int c = 0; c < 16; ++c) oacc[c] = (v8f){0.f,0.f,0.f,0.f,0.f,0.f,0.f,0.f};
  float m_i[8], l_i[8];
#pragma unroll
  for (int r = 0; r < 8; ++r) { m_i[r] = NEGINF; l_i[r] = 0.f; }

  int ktiles = causal ? (4 * (int)blockIdx.y + 4) : (SEQ / 32);

  for (int kt = 0; kt < ktiles; ++kt) {
    const int kk0 = kt * 32;
    __syncthreads();
    {                                         // stage K tile 32x256
      int r  = tid >> 3;
      int c0 = (tid & 7) * 32;
#pragma unroll
      for (int j = 0; j < 4; ++j) {
        uint4 v = *(const uint4*)(Kb + (size_t)(kk0 + r) * HDIM + c0 + j * 8);
        *(uint4*)(&Ks[r][c0 + j * 8]) = v;
      }
    }
    {                                         // stage V tile 256x32
      int r = tid;
#pragma unroll
      for (int j = 0; j < 4; ++j) {
        uint4 v = *(const uint4*)(Vb + (size_t)r * SEQ + kk0 + j * 8);
        *(uint4*)(&Vs[r][j * 8]) = v;
      }
    }
    __syncthreads();

    // ---- scores: 16 q rows x 32 keys ----
    v8f s0 = (v8f){0.f,0.f,0.f,0.f,0.f,0.f,0.f,0.f};
    v8f s1 = s0;
#pragma unroll
    for (int dk = 0; dk < HDIM; dk += 32) {
      Frag af, b0, b1;
      int qm = qw + lo;
      af.q[0] = *(const uint4*)(Qb + (size_t)qm * HDIM + dk + kb);
      af.q[1] = *(const uint4*)(Qb + (size_t)qm * HDIM + dk + 16 + kb);
      b0.q[0] = *(const uint4*)(&Ks[lo][dk + kb]);
      b0.q[1] = *(const uint4*)(&Ks[lo][dk + 16 + kb]);
      b1.q[0] = *(const uint4*)(&Ks[16 + lo][dk + kb]);
      b1.q[1] = *(const uint4*)(&Ks[16 + lo][dk + 16 + kb]);
      s0 = wmma_bf16(af.v, b0.v, s0);
      s1 = wmma_bf16(af.v, b1.v, s1);
    }

    // ---- online softmax (rows live across a 16-lane half-wave) ----
    const float scale = 0.0625f;              // 1/sqrt(256)
    float alph[8];
#pragma unroll
    for (int r = 0; r < 8; ++r) {
      int qi = qw + hi * 8 + r;
      float a0 = s0[r] * scale;
      float a1 = s1[r] * scale;
      if (causal) {
        if (kk0 + lo > qi)      a0 = NEGINF;
        if (kk0 + 16 + lo > qi) a1 = NEGINF;
      }
      float mx = fmaxf(a0, a1);
#pragma unroll
      for (int d = 1; d < 16; d <<= 1) mx = fmaxf(mx, __shfl_xor(mx, d, 32));
      float mnew = fmaxf(m_i[r], mx);
      float e0 = __expf(a0 - mnew);
      float e1 = __expf(a1 - mnew);
      float al = __expf(m_i[r] - mnew);
      float sum = e0 + e1;
#pragma unroll
      for (int d = 1; d < 16; d <<= 1) sum += __shfl_xor(sum, d, 32);
      l_i[r] = l_i[r] * al + sum;
      m_i[r] = mnew;
      alph[r] = al;
      Ps[wid][hi * 8 + r][lo]      = f2bf(e0);
      Ps[wid][hi * 8 + r][16 + lo] = f2bf(e1);
    }
#pragma unroll
    for (int c = 0; c < 16; ++c)
#pragma unroll
      for (int r = 0; r < 8; ++r) oacc[c][r] *= alph[r];

    // wave-private LDS round trip: C-layout -> A-layout
    asm volatile("s_wait_dscnt 0" ::: "memory");

    Frag pf;
    pf.q[0] = *(const uint4*)(&Ps[wid][lo][kb]);
    pf.q[1] = *(const uint4*)(&Ps[wid][lo][16 + kb]);

    // ---- att @ V across HDIM in chunks of 16 ----
#pragma unroll
    for (int c = 0; c < 16; ++c) {
      Frag vf;
      int n = c * 16 + lo;
      vf.q[0] = *(const uint4*)(&Vs[n][kb]);
      vf.q[1] = *(const uint4*)(&Vs[n][16 + kb]);
      oacc[c] = wmma_bf16(pf.v, vf.v, oacc[c]);
    }
  }

  // ---- normalize and write concat-head output ----
#pragma unroll
  for (int c = 0; c < 16; ++c) {
#pragma unroll
    for (int r = 0; r < 8; ++r) {
      int qi = qw + hi * 8 + r;
      int dh = c * 16 + lo;
      float v = oacc[c][r] / l_i[r];
      size_t idx = ((size_t)b * SEQ + qi) * DMODEL + h * HDIM + dh;
      if (Of) Of[idx] = v;
      if (Ob) Ob[idx] = f2bf(v);
    }
  }
}

// ---------------------------------------------------------------------------
// Residual + LayerNorm: out = LN(a + r) * g + b.  One wave per token (D=512).
// ---------------------------------------------------------------------------
__global__ __launch_bounds__(256) void ln_kernel(
    const float* __restrict__ A, const float* __restrict__ R,
    const float* __restrict__ g, const float* __restrict__ be,
    float* __restrict__ outf, unsigned short* __restrict__ outb) {
  int token = blockIdx.x * 8 + (threadIdx.x >> 5);
  int lane  = threadIdx.x & 31;
  const float* a = A + (size_t)token * DMODEL;
  const float* r = R + (size_t)token * DMODEL;
  float vals[16];
  float sum = 0.f;
#pragma unroll
  for (int j = 0; j < 16; ++j) {
    int c = j * 32 + lane;
    float v = a[c] + r[c];
    vals[j] = v;
    sum += v;
  }
#pragma unroll
  for (int d = 1; d < 32; d <<= 1) sum += __shfl_xor(sum, d, 32);
  float mean = sum * (1.f / DMODEL);
  float vs = 0.f;
#pragma unroll
  for (int j = 0; j < 16; ++j) { float t = vals[j] - mean; vs += t * t; }
#pragma unroll
  for (int d = 1; d < 32; d <<= 1) vs += __shfl_xor(vs, d, 32);
  float inv = rsqrtf(vs * (1.f / DMODEL) + 1e-5f);
#pragma unroll
  for (int j = 0; j < 16; ++j) {
    int c = j * 32 + lane;
    float o = (vals[j] - mean) * inv * g[c] + be[c];
    if (outf) outf[(size_t)token * DMODEL + c] = o;
    if (outb) outb[(size_t)token * DMODEL + c] = f2bf(o);
  }
}

// ---------------------------------------------------------------------------
// Host orchestration
// ---------------------------------------------------------------------------
typedef unsigned short u16;

static void convT(const float* src, u16* dst, int K, int N, hipStream_t s) {
  dim3 g(K / 32, N / 32);
  cvtT_kernel<<<g, 256, 0, s>>>(src, dst, K, N);
}

static void gemm(const u16* A, const u16* BT, const float* bias, float* C,
                 u16* Cbf, u16* Qo, u16* Ko, u16* Vo, int M, int N, int K,
                 int relu, hipStream_t s) {
  dim3 g(M / 128, N / 128);
  gemm_bf16_kernel<<<g, 256, 0, s>>>(A, BT, bias, C, Cbf, Qo, Ko, Vo, M, N, K, relu);
}

extern "C" void kernel_launch(void* const* d_in, const int* in_sizes, int n_in,
                              void* d_out, int out_size, void* d_ws, size_t ws_size,
                              hipStream_t stream) {
  (void)in_sizes; (void)n_in; (void)out_size; (void)ws_size;
  // ---- input layout (setup_inputs dict order, recursively flattened) ----
  const float* enc_tok = (const float*)d_in[0];
  const float* enc_pos = (const float*)d_in[1];
  const float* dec_tok = (const float*)d_in[2];
  const float* dec_pos = (const float*)d_in[3];
  const float* w_qkv   = (const float*)d_in[4];
  const float* b_qkv   = (const float*)d_in[5];
  // enc layer l at 6+10l: wo,bo,ln1.g,ln1.b,w1,b1,w2,b2,ln2.g,ln2.b
  // dec layer l at 36+14l: wo1,bo1,ln1.g,ln1.b,wo2,bo2,ln2.g,ln2.b,w1,b1,w2,b2,ln3.g,ln3.b
  const float* w_out   = (const float*)d_in[78];
  const float* b_out   = (const float*)d_in[79];
  const int* enc_ids   = (const int*)d_in[80];
  const int* dec_ids   = (const int*)d_in[81];

  // ---- bump allocator over workspace ----
  char* wp = (char*)d_ws;
  auto alloc = [&](size_t bytes) -> char* {
    char* r = wp;
    wp += (bytes + 255) & ~(size_t)255;
    return r;
  };
  const size_t TOKD = (size_t)MTOK * DMODEL;      // 4,194,304
  // all weight copies are TRANSPOSED: BT[N][K]
  u16* wqkv_b = (u16*)alloc((size_t)3 * DMODEL * DMODEL * 2);
  u16* wout_b = (u16*)alloc((size_t)16000 * DMODEL * 2);
  u16 *ewo[3], *ew1[3], *ew2[3], *dwo1[3], *dwo2[3], *dw1[3], *dw2[3];
  for (int l = 0; l < 3; ++l) {
    ewo[l]  = (u16*)alloc((size_t)DMODEL * DMODEL * 2);
    ew1[l]  = (u16*)alloc((size_t)FFDIM * DMODEL * 2);
    ew2[l]  = (u16*)alloc((size_t)DMODEL * FFDIM * 2);
    dwo1[l] = (u16*)alloc((size_t)DMODEL * DMODEL * 2);
    dwo2[l] = (u16*)alloc((size_t)DMODEL * DMODEL * 2);
    dw1[l]  = (u16*)alloc((size_t)FFDIM * DMODEL * 2);
    dw2[l]  = (u16*)alloc((size_t)DMODEL * FFDIM * 2);
  }
  float* x_f  = (float*)alloc(TOKD * 4);  u16* x_b  = (u16*)alloc(TOKD * 2);
  float* y_f  = (float*)alloc(TOKD * 4);  u16* y_b  = (u16*)alloc(TOKD * 2);
  float* t1_f = (float*)alloc(TOKD * 4);  u16* t1_b = (u16*)alloc(TOKD * 2);
  float* t2_f = (float*)alloc(TOKD * 4);  u16* t2_b = (u16*)alloc(TOKD * 2);
  float* pr_f  = (float*)alloc(TOKD * 4);
  float* pr2_f = (float*)alloc(TOKD * 4);
  u16* att_b = (u16*)alloc(TOKD * 2);
  u16* h_b   = (u16*)alloc((size_t)MTOK * FFDIM * 2);
  u16* q_b   = (u16*)alloc(TOKD * 2);
  u16* k_b   = (u16*)alloc(TOKD * 2);
  u16* v_b   = (u16*)alloc(TOKD * 2);     // [B,H,DH,S]
  u16* qf_b  = (u16*)alloc(TOKD * 2);
  u16* kf_b  = (u16*)alloc(TOKD * 2);

  // ---- weight conversion + transpose (every call; determinism) ----
  convT(w_qkv, wqkv_b, DMODEL, 3 * DMODEL, stream);
  convT(w_out, wout_b, DMODEL, 16000, stream);
  for (int l = 0; l < 3; ++l) {
    const float* const* E = (const float* const*)&d_in[6 + 10 * l];
    convT(E[0], ewo[l], DMODEL, DMODEL, stream);
    convT(E[4], ew1[l], DMODEL, FFDIM, stream);
    convT(E[6], ew2[l], FFDIM, DMODEL, stream);
    const float* const* D = (const float* const*)&d_in[36 + 14 * l];
    convT(D[0], dwo1[l], DMODEL, DMODEL, stream);
    convT(D[4], dwo2[l], DMODEL, DMODEL, stream);
    convT(D[8], dw1[l], DMODEL, FFDIM, stream);
    convT(D[10], dw2[l], FFDIM, DMODEL, stream);
  }

  const dim3 agrid(BATCH * NHEADS, SEQ / 128);

  // ---- encoder ----
  embed_kernel<<<MTOK, 256, 0, stream>>>(enc_tok, enc_pos, enc_ids, x_f, x_b);
  for (int l = 0; l < 3; ++l) {
    const float* const* E = (const float* const*)&d_in[6 + 10 * l];
    gemm(x_b, wqkv_b, b_qkv, nullptr, nullptr, q_b, k_b, v_b,
         MTOK, 3 * DMODEL, DMODEL, 0, stream);
    attn_kernel<<<agrid, 256, 0, stream>>>(q_b, k_b, v_b, nullptr, att_b, 0);
    gemm(att_b, ewo[l], E[1], pr_f, nullptr, nullptr, nullptr, nullptr,
         MTOK, DMODEL, DMODEL, 0, stream);
    ln_kernel<<<MTOK / 8, 256, 0, stream>>>(pr_f, x_f, E[2], E[3], t1_f, t1_b);
    gemm(t1_b, ew1[l], E[5], nullptr, h_b, nullptr, nullptr, nullptr,
         MTOK, FFDIM, DMODEL, 1, stream);
    gemm(h_b, ew2[l], E[7], pr_f, nullptr, nullptr, nullptr, nullptr,
         MTOK, DMODEL, FFDIM, 0, stream);
    ln_kernel<<<MTOK / 8, 256, 0, stream>>>(pr_f, t1_f, E[8], E[9], x_f, x_b);
  }

  // ---- decoder prelude: qf, kf from final encoder state ----
  gemm(x_b, wqkv_b, b_qkv, nullptr, nullptr, qf_b, kf_b, v_b,
       MTOK, 3 * DMODEL, DMODEL, 0, stream);
  embed_kernel<<<MTOK, 256, 0, stream>>>(dec_tok, dec_pos, dec_ids, y_f, y_b);

  // ---- decoder ----
  for (int l = 0; l < 3; ++l) {
    const float* const* D = (const float* const*)&d_in[36 + 14 * l];
    gemm(y_b, wqkv_b, b_qkv, nullptr, nullptr, q_b, k_b, v_b,
         MTOK, 3 * DMODEL, DMODEL, 0, stream);           // vd = v_b
    attn_kernel<<<agrid, 256, 0, stream>>>(qf_b, kf_b, v_b, nullptr, att_b, 1);
    gemm(att_b, dwo1[l], D[1], pr_f, nullptr, nullptr, nullptr, nullptr,
         MTOK, DMODEL, DMODEL, 0, stream);
    ln_kernel<<<MTOK / 8, 256, 0, stream>>>(pr_f, y_f, D[2], D[3], t1_f, t1_b); // y1
    attn_kernel<<<agrid, 256, 0, stream>>>(qf_b, kf_b, v_b, nullptr, att_b, 0);
    gemm(att_b, dwo2[l], D[5], pr2_f, nullptr, nullptr, nullptr, nullptr,
         MTOK, DMODEL, DMODEL, 0, stream);
    ln_kernel<<<MTOK / 8, 256, 0, stream>>>(pr2_f, t1_f, D[6], D[7], t2_f, t2_b); // y2
    gemm(t2_b, dw1[l], D[9], nullptr, h_b, nullptr, nullptr, nullptr,
         MTOK, FFDIM, DMODEL, 1, stream);
    gemm(h_b, dw2[l], D[11], pr_f, nullptr, nullptr, nullptr, nullptr,
         MTOK, DMODEL, FFDIM, 0, stream);
    ln_kernel<<<MTOK / 8, 256, 0, stream>>>(pr_f, t2_f, D[12], D[13], y_f, y_b);
  }

  // ---- logits: [8192,16000] f32 straight to d_out ----
  gemm(y_b, wout_b, b_out, (float*)d_out, nullptr, nullptr, nullptr, nullptr,
       MTOK, 16000, DMODEL, 0, stream);
}
